// LSTM_cell_31439160607294
// MI455X (gfx1250) — matrix operand order repaired and checked by reading, not compiled, not run on hardware
//
#include <hip/hip_runtime.h>
#include <hip/hip_bf16.h>

// ---------------------------------------------------------------------------
// Problem constants (from the reference): B=64, T=512, I=1024, H=1024
// ---------------------------------------------------------------------------
#define B_   64
#define T_   512
#define I_   1024
#define H_   1024
#define BT_  (B_ * T_)          // 32768 rows of the big GEMM

typedef __attribute__((ext_vector_type(16))) __bf16 v16bf;
typedef __attribute__((ext_vector_type(8)))  __bf16 v8bf;
typedef __attribute__((ext_vector_type(8)))  float  v8f;

// ---------------------------------------------------------------------------
// fp32 -> bf16 (round to nearest even)
// ---------------------------------------------------------------------------
__device__ __forceinline__ unsigned int f2bf(float f) {
  unsigned int u = __float_as_uint(f);
  u += 0x7FFFu + ((u >> 16) & 1u);
  return u >> 16;
}

// Each thread converts 8 consecutive floats -> 8 bf16 (one 16B store).
__global__ __launch_bounds__(256) void cvt_bf16_kernel(
    const float* __restrict__ src, unsigned short* __restrict__ dst, int n8) {
  int i = blockIdx.x * blockDim.x + threadIdx.x;
  if (i >= n8) return;
  const float4* s = (const float4*)src;
  float4 a = s[2 * i];
  float4 b = s[2 * i + 1];
  uint4 o;
  o.x = f2bf(a.x) | (f2bf(a.y) << 16);
  o.y = f2bf(a.z) | (f2bf(a.w) << 16);
  o.z = f2bf(b.x) | (f2bf(b.y) << 16);
  o.w = f2bf(b.z) | (f2bf(b.w) << 16);
  ((uint4*)dst)[i] = o;
}

// ---------------------------------------------------------------------------
// rT[b*H + n] = bias[n] + rbias[n] + dot(h0[b,:], R[n,:])   (tiny GEMM, fp32)
// ---------------------------------------------------------------------------
__global__ __launch_bounds__(256) void recur_kernel(
    const float* __restrict__ h0, const float* __restrict__ R,
    const float* __restrict__ bias, const float* __restrict__ rbias,
    float* __restrict__ rT) {
  int idx = blockIdx.x * blockDim.x + threadIdx.x;   // [0, B*H)
  int b = idx >> 10;            // / H_
  int n = idx & (H_ - 1);
  const float4* hp = (const float4*)(h0 + (size_t)b * H_);
  const float4* rp = (const float4*)(R + (size_t)n * H_);
  float acc = 0.f;
#pragma unroll 4
  for (int k = 0; k < H_ / 4; ++k) {
    float4 hv = hp[k];
    float4 rv = rp[k];
    acc += hv.x * rv.x + hv.y * rv.y + hv.z * rv.z + hv.w * rv.w;
  }
  rT[idx] = acc + bias[n] + rbias[n];
}

// ---------------------------------------------------------------------------
// Main fused GEMM + LSTM epilogue.
// Grid: (BT/128, H/64). Block: 256 threads = 8 wave32, arranged 4(M) x 2(N).
// Each wave: 32x32 output tile = 2x2 WMMA subtiles, x3 gates (f, i, g).
// A (x, bf16) fragment 16x32: lane<16 holds M=lane, K={0..15};
//                             lane>=16 holds M=lane-16, K={16..31}.
// B (W row-major = W^T col, bf16) fragment 32x16: lane l holds N=l&15,
//   K = (l<16?0:16) + {0..15} -> 16 contiguous bf16 of W[n, :].
// ---------------------------------------------------------------------------
__global__ __launch_bounds__(256) void lstm_gemm_kernel(
    const __bf16* __restrict__ Xb,
    const __bf16* __restrict__ Wfb,
    const __bf16* __restrict__ Wib,
    const __bf16* __restrict__ Wgb,
    const float* __restrict__ rTf,
    const float* __restrict__ rTi,
    const float* __restrict__ rTg,
    const float* __restrict__ c0,
    float* __restrict__ out) {
  const int lane = threadIdx.x & 31;
  const int wave = threadIdx.x >> 5;
  const int wm = wave & 3;        // 0..3  (M direction)
  const int wn = wave >> 2;       // 0..1  (N direction)
  const int mbase = blockIdx.x * 128 + wm * 32;
  const int nbase = blockIdx.y * 64 + wn * 32;

  const int l16 = lane & 15;
  const int hi  = lane >> 4;      // 0 or 1 (lane half)

  v8f accf[2][2] = {{{0.f}, {0.f}}, {{0.f}, {0.f}}};
  v8f acci[2][2] = {{{0.f}, {0.f}}, {{0.f}, {0.f}}};
  v8f accg[2][2] = {{{0.f}, {0.f}}, {{0.f}, {0.f}}};

  // Per-lane base pointers.
  const __bf16* arow[2];
  const __bf16* brf[2];
  const __bf16* bri[2];
  const __bf16* brg[2];
#pragma unroll
  for (int mi = 0; mi < 2; ++mi)
    arow[mi] = Xb + (size_t)(mbase + mi * 16 + l16) * I_;
#pragma unroll
  for (int ni = 0; ni < 2; ++ni) {
    size_t nrow = (size_t)(nbase + ni * 16 + l16) * I_;
    brf[ni] = Wfb + nrow;
    bri[ni] = Wib + nrow;
    brg[ni] = Wgb + nrow;
  }
  const int a_koff = hi * 16;     // A: lane half selects K sub-block of 16
  const int b_koff = hi * 16;     // B: lane half selects K sub-block of 16

#pragma unroll 2
  for (int k = 0; k < I_; k += 32) {
    v16bf a[2], bf_[3][2];
#pragma unroll
    for (int mi = 0; mi < 2; ++mi) {
      v8bf lo = *(const v8bf*)(arow[mi] + k + a_koff);
      v8bf hp = *(const v8bf*)(arow[mi] + k + a_koff + 8);
      a[mi] = __builtin_shufflevector(lo, hp, 0, 1, 2, 3, 4, 5, 6, 7,
                                      8, 9, 10, 11, 12, 13, 14, 15);
      __builtin_prefetch(arow[mi] + k + 64, 0, 3);   // global_prefetch_b8
    }
#pragma unroll
    for (int ni = 0; ni < 2; ++ni) {
      bf_[0][ni] = *(const v16bf*)(brf[ni] + k + b_koff);
      bf_[1][ni] = *(const v16bf*)(bri[ni] + k + b_koff);
      bf_[2][ni] = *(const v16bf*)(brg[ni] + k + b_koff);
    }
#pragma unroll
    for (int mi = 0; mi < 2; ++mi)
#pragma unroll
      for (int ni = 0; ni < 2; ++ni) {
        accf[mi][ni] = __builtin_amdgcn_wmma_f32_16x16x32_bf16(
            false, a[mi], false, bf_[0][ni], (short)0, accf[mi][ni], false, false);
        acci[mi][ni] = __builtin_amdgcn_wmma_f32_16x16x32_bf16(
            false, a[mi], false, bf_[1][ni], (short)0, acci[mi][ni], false, false);
        accg[mi][ni] = __builtin_amdgcn_wmma_f32_16x16x32_bf16(
            false, a[mi], false, bf_[2][ni], (short)0, accg[mi][ni], false, false);
      }
  }

  // Epilogue. C/D layout: lane holds fixed N = n0 + (lane&15); VGPR j holds
  // M = m0 + (lane>=16 ? 8 : 0) + j. Any aligned 8-row block never crosses a
  // T=512 boundary, so the batch index b is constant across j.
#pragma unroll
  for (int mi = 0; mi < 2; ++mi) {
#pragma unroll
    for (int ni = 0; ni < 2; ++ni) {
      const int n = nbase + ni * 16 + l16;
      const int m0 = mbase + mi * 16 + hi * 8;      // multiple of 8
      const int bb = m0 >> 9;                       // / T_
      const int t0 = m0 & (T_ - 1);
      const size_t bh = (size_t)bb * H_ + n;
      const float rf = rTf[bh];
      const float ri = rTi[bh];
      const float rg = rTg[bh];
      const float c0v = c0[bh];
      float* ocol = out + (size_t)n * BT_ + (size_t)t0 * B_ + bb;
#pragma unroll
      for (int j = 0; j < 8; ++j) {
        float pf = accf[mi][ni][j] + rf;
        float pi = acci[mi][ni][j] + ri;
        float pg = accg[mi][ni][j] + rg;
        float fg = 1.f / (1.f + __expf(-pf));
        float ig = 1.f / (1.f + __expf(-pi));
        float gg = tanhf(pg);
        float cc = fg * c0v + ig * gg;
        float hh = tanhf(cc);
        ocol[(size_t)j * B_] = hh;                  // hidden_seq[n, t*B + b]
        if (t0 + j == T_ - 1) {                     // last timestep extras
          out[(size_t)H_ * BT_ + bh] = hh;                         // h_last
          out[(size_t)H_ * BT_ + (size_t)B_ * H_ + bh] = cc;       // c_last
        }
      }
    }
  }
}

// ---------------------------------------------------------------------------
// Launch
// ---------------------------------------------------------------------------
extern "C" void kernel_launch(void* const* d_in, const int* in_sizes, int n_in,
                              void* d_out, int out_size, void* d_ws, size_t ws_size,
                              hipStream_t stream) {
  const float* x   = (const float*)d_in[0];
  const float* h0  = (const float*)d_in[1];
  const float* c0  = (const float*)d_in[2];
  const float* Wf  = (const float*)d_in[3];
  const float* bf  = (const float*)d_in[4];
  const float* Rf  = (const float*)d_in[5];
  const float* rbf = (const float*)d_in[6];
  const float* Wi  = (const float*)d_in[7];
  const float* bi  = (const float*)d_in[8];
  const float* Ri  = (const float*)d_in[9];
  const float* rbi = (const float*)d_in[10];
  const float* Wg  = (const float*)d_in[11];
  const float* bg  = (const float*)d_in[12];
  const float* Rg  = (const float*)d_in[13];
  const float* rbg = (const float*)d_in[14];
  // d_in[15..18] (output gate) are computed-but-unused in the reference.
  float* out = (float*)d_out;

  // Workspace layout (bytes): Xb 64MB | Wf 2MB | Wi 2MB | Wg 2MB | rT 768KB
  unsigned short* Xb  = (unsigned short*)d_ws;
  unsigned short* Wfb = Xb + (size_t)BT_ * I_;
  unsigned short* Wib = Wfb + (size_t)H_ * I_;
  unsigned short* Wgb = Wib + (size_t)H_ * I_;
  float* rT  = (float*)(Wgb + (size_t)H_ * I_);
  float* rTf = rT;
  float* rTi = rT + (size_t)B_ * H_;
  float* rTg = rT + 2 * (size_t)B_ * H_;

  // 1) fp32 -> bf16 conversions (memory-bound, one pass).
  {
    int n8 = BT_ * I_ / 8;                       // x: 33.5M elements
    cvt_bf16_kernel<<<(n8 + 255) / 256, 256, 0, stream>>>(x, Xb, n8);
    int w8 = H_ * I_ / 8;                        // each W: 1M elements
    cvt_bf16_kernel<<<(w8 + 255) / 256, 256, 0, stream>>>(Wf, Wfb, w8);
    cvt_bf16_kernel<<<(w8 + 255) / 256, 256, 0, stream>>>(Wi, Wib, w8);
    cvt_bf16_kernel<<<(w8 + 255) / 256, 256, 0, stream>>>(Wg, Wgb, w8);
  }

  // 2) Recurrent terms + both biases folded: rT_g = b_g + rb_g + h0 @ R_g^T.
  {
    int nthreads = B_ * H_;                      // 65536
    recur_kernel<<<nthreads / 256, 256, 0, stream>>>(h0, Rf, bf, rbf, rTf);
    recur_kernel<<<nthreads / 256, 256, 0, stream>>>(h0, Ri, bi, rbi, rTi);
    recur_kernel<<<nthreads / 256, 256, 0, stream>>>(h0, Rg, bg, rbg, rTg);
  }

  // 3) Fused 3-gate WMMA GEMM + LSTM epilogue + transposed store.
  dim3 grid(BT_ / 128, H_ / 64);                 // 256 x 16 workgroups
  lstm_gemm_kernel<<<grid, 256, 0, stream>>>(Xb ? (const __bf16*)Xb : nullptr,
                                             (const __bf16*)Wfb,
                                             (const __bf16*)Wib,
                                             (const __bf16*)Wgb,
                                             rTf, rTi, rTg, c0, out);
}
